// VectorQuantizer_21964462751904
// MI455X (gfx1250) — matrix operand order repaired and checked by reading, not compiled, not run on hardware
//
#include <hip/hip_runtime.h>

typedef __attribute__((ext_vector_type(16))) _Float16 v16h;
typedef __attribute__((ext_vector_type(8)))  _Float16 v8h;
typedef __attribute__((ext_vector_type(8)))  float    v8f;

#define KC         1024   // codebook size
#define DIM        64     // C*W
#define PDIM       8192   // patches per batch
#define LDS_STRIDE 72     // halves per codebook row in LDS (64 + 8 pad, 16B aligned)

// ---------------------------------------------------------------------------
// Prep: f32 codebook -> f16 copy in workspace + per-code squared norms (f32).
// ---------------------------------------------------------------------------
__global__ __launch_bounds__(32)
void vq_prep(const float* __restrict__ cb, _Float16* __restrict__ cb16,
             float* __restrict__ csq) {
  const int j    = blockIdx.x;
  const int lane = threadIdx.x;
  float v0 = cb[j * DIM + lane];
  float v1 = cb[j * DIM + lane + 32];
  cb16[j * DIM + lane]      = (_Float16)v0;
  cb16[j * DIM + lane + 32] = (_Float16)v1;
  float s = v0 * v0 + v1 * v1;
#pragma unroll
  for (int m = 16; m >= 1; m >>= 1) s += __shfl_xor(s, m, 32);
  if (lane == 0) csq[j] = s;
}

// B tile: one 16-code tile of the codebook, both 32-deep K chunks.
struct BTile {
  union { v16h v; v8h h[2]; } c0, c1;
};

__device__ __forceinline__ void load_btile(const _Float16* lds_cb, int code,
                                           int halfsel, BTile& bt) {
  const _Float16* bp = &lds_cb[code * LDS_STRIDE + halfsel * 8];
  bt.c0.h[0] = *(const v8h*)(bp);          // K =      halfsel*8 + 0..7
  bt.c0.h[1] = *(const v8h*)(bp + 16);     // K = 16 + halfsel*8 + 0..7
  bt.c1.h[0] = *(const v8h*)(bp + 32);     // K = 32 + halfsel*8 + 0..7
  bt.c1.h[1] = *(const v8h*)(bp + 48);
}

// One code-tile step for two token tiles (A and B fragments).
__device__ __forceinline__ void vq_step(const v16h& a0A, const v16h& a1A,
                                        const v16h& a0B, const v16h& a1B,
                                        const BTile& bt, float cs, int code,
                                        float (&bdA)[8], int (&biA)[8],
                                        float (&bdB)[8], int (&biB)[8]) {
  v8f accA = {};
  accA = __builtin_amdgcn_wmma_f32_16x16x32_f16(false, a0A, false, bt.c0.v,
                                                (short)0, accA, false, false);
  accA = __builtin_amdgcn_wmma_f32_16x16x32_f16(false, a1A, false, bt.c1.v,
                                                (short)0, accA, false, false);
  v8f accB = {};
  accB = __builtin_amdgcn_wmma_f32_16x16x32_f16(false, a0B, false, bt.c0.v,
                                                (short)0, accB, false, false);
  accB = __builtin_amdgcn_wmma_f32_16x16x32_f16(false, a1B, false, bt.c1.v,
                                                (short)0, accB, false, false);
#pragma unroll
  for (int g = 0; g < 8; ++g) {
    float dA = __builtin_fmaf(-2.0f, accA[g], cs);
    if (dA < bdA[g]) { bdA[g] = dA; biA[g] = code; }
    float dB = __builtin_fmaf(-2.0f, accB[g], cs);
    if (dB < bdB[g]) { bdB[g] = dB; biB[g] = code; }
  }
}

// ---------------------------------------------------------------------------
// Main fused kernel: WMMA GEMM + argmin + codebook gather/store.
// grid = 128 blocks x 256 threads (8 waves); block handles 1024 tokens.
// ---------------------------------------------------------------------------
__global__ __launch_bounds__(256)
void vq_main(const float* __restrict__ x, const float* __restrict__ cb_f32,
             const _Float16* __restrict__ cb16, const float* __restrict__ csq,
             float* __restrict__ out) {
  extern __shared__ char smem[];
  _Float16* lds_cb  = (_Float16*)smem;                               // 1024*72 halves
  float*    lds_csq = (float*)(smem + KC * LDS_STRIDE * 2);          // 1024 f32
  int*      lds_idx = (int*)(smem + KC * LDS_STRIDE * 2 + KC * 4);   // 8 waves * 32

  const int tid = threadIdx.x;

  // ---- Async fill of the f16 codebook into padded LDS (ASYNCcnt path) ----
  {
    const uint4* src = (const uint4*)cb16;          // 8 halves per uint4
    for (int i = tid; i < KC * DIM / 8; i += 256) {
      const int row  = i >> 3;
      const int col8 = (i & 7) * 8;
      unsigned lds_off = (unsigned)((row * LDS_STRIDE + col8) * 2);
      const uint4* gp = src + i;
      asm volatile("global_load_async_to_lds_b128 %0, %1, off"
                   :: "v"(lds_off), "v"(gp) : "memory");
    }
    for (int i = tid; i < KC; i += 256) lds_csq[i] = csq[i];
    asm volatile("s_wait_asynccnt 0" ::: "memory");
  }
  __syncthreads();

  const int lane    = tid & 31;
  const int wave    = tid >> 5;      // 0..7
  const int halfsel = lane >> 4;     // 0 | 1
  const int lmod    = lane & 15;

  const int b     = blockIdx.x >> 3;
  const int p_blk = (blockIdx.x & 7) * 1024;

  const float* xb   = x   + (size_t)b * DIM * PDIM;
  float*       outb = out + (size_t)b * DIM * PDIM;

  for (int t = 0; t < 4; ++t) {                    // 4 pairs of 16-token tiles
    const int p0A = p_blk + wave * 128 + t * 32;
    const int p0B = p0A + 16;

    // ---- Load two A tiles (f32 -> f16), streaming (non-temporal) ----
    const float* xpA = xb + p0A + lmod;
    const float* xpB = xb + p0B + lmod;
    v16h a0A, a1A, a0B, a1B;
#pragma unroll
    for (int i = 0; i < 16; ++i) {
      const int d0 = halfsel * 8 + (i >> 3) * 16 + (i & 7);     // 0..31
      a0A[i] = (_Float16)__builtin_nontemporal_load(&xpA[(size_t)d0 * PDIM]);
      a1A[i] = (_Float16)__builtin_nontemporal_load(&xpA[(size_t)(d0 + 32) * PDIM]);
      a0B[i] = (_Float16)__builtin_nontemporal_load(&xpB[(size_t)d0 * PDIM]);
      a1B[i] = (_Float16)__builtin_nontemporal_load(&xpB[(size_t)(d0 + 32) * PDIM]);
    }

    float bdA[8], bdB[8];
    int   biA[8], biB[8];
#pragma unroll
    for (int g = 0; g < 8; ++g) {
      bdA[g] = 3.4e38f; biA[g] = 0;
      bdB[g] = 3.4e38f; biB[g] = 0;
    }

    // ---- Sweep the codebook, B tiles double-buffered in registers ----
    BTile bt0, bt1;
    load_btile(lds_cb, lmod, halfsel, bt0);                    // prime tile 0
    for (int jt = 0; jt < 64; jt += 2) {
      load_btile(lds_cb, (jt + 1) * 16 + lmod, halfsel, bt1);  // prefetch jt+1
      vq_step(a0A, a1A, a0B, a1B, bt0, lds_csq[jt * 16 + lmod],
              jt * 16 + lmod, bdA, biA, bdB, biB);
      if (jt + 2 < 64)
        load_btile(lds_cb, (jt + 2) * 16 + lmod, halfsel, bt0); // prefetch jt+2
      vq_step(a0A, a1A, a0B, a1B, bt1, lds_csq[(jt + 1) * 16 + lmod],
              (jt + 1) * 16 + lmod, bdA, biA, bdB, biB);
    }

    // ---- Argmin reduction across the 16 lanes of the N dimension ----
#pragma unroll
    for (int m = 8; m >= 1; m >>= 1) {
#pragma unroll
      for (int g = 0; g < 8; ++g) {
        float odA = __shfl_xor(bdA[g], m, 32);
        int   oiA = __shfl_xor(biA[g], m, 32);
        if (odA < bdA[g] || (odA == bdA[g] && oiA < biA[g])) { bdA[g] = odA; biA[g] = oiA; }
        float odB = __shfl_xor(bdB[g], m, 32);
        int   oiB = __shfl_xor(biB[g], m, 32);
        if (odB < bdB[g] || (odB == bdB[g] && oiB < biB[g])) { bdB[g] = odB; biB[g] = oiB; }
      }
    }

    // ---- Stage winning indices per wave (same-wave LDS is in-order) ----
    if (lmod == 0) {
#pragma unroll
      for (int g = 0; g < 8; ++g) {
        lds_idx[wave * 32 + halfsel * 8 + g]      = biA[g];
        lds_idx[wave * 32 + 16 + halfsel * 8 + g] = biB[g];
      }
    }
    const int idxA = lds_idx[wave * 32 + lmod];
    const int idxB = lds_idx[wave * 32 + 16 + lmod];

    // ---- Gather winning f32 codebook rows + coalesced non-temporal store ----
#pragma unroll
    for (int cc0 = 0; cc0 < 8; ++cc0) {
      const int c = cc0 * 8 + halfsel * 4;
      const float4 vA = *(const float4*)&cb_f32[(size_t)idxA * DIM + c];
      float* opA = outb + (size_t)c * PDIM + p0A + lmod;
      __builtin_nontemporal_store(vA.x, &opA[0 * PDIM]);
      __builtin_nontemporal_store(vA.y, &opA[1 * PDIM]);
      __builtin_nontemporal_store(vA.z, &opA[2 * PDIM]);
      __builtin_nontemporal_store(vA.w, &opA[3 * PDIM]);
      const float4 vB = *(const float4*)&cb_f32[(size_t)idxB * DIM + c];
      float* opB = outb + (size_t)c * PDIM + p0B + lmod;
      __builtin_nontemporal_store(vB.x, &opB[0 * PDIM]);
      __builtin_nontemporal_store(vB.y, &opB[1 * PDIM]);
      __builtin_nontemporal_store(vB.z, &opB[2 * PDIM]);
      __builtin_nontemporal_store(vB.w, &opB[3 * PDIM]);
    }
  }
}

// ---------------------------------------------------------------------------
extern "C" void kernel_launch(void* const* d_in, const int* in_sizes, int n_in,
                              void* d_out, int out_size, void* d_ws, size_t ws_size,
                              hipStream_t stream) {
  (void)in_sizes; (void)n_in; (void)out_size; (void)ws_size;
  const float* x  = (const float*)d_in[0];   // (16, 64, 1, 8192) f32
  const float* cb = (const float*)d_in[1];   // (1024, 64) f32
  float* out = (float*)d_out;                // (16, 64, 1, 8192) f32

  _Float16* cb16 = (_Float16*)d_ws;                              // 128 KB
  float*    csq  = (float*)((char*)d_ws + (size_t)KC * DIM * 2); // 4 KB

  vq_prep<<<KC, 32, 0, stream>>>(cb, cb16, csq);

  const size_t smem = (size_t)KC * LDS_STRIDE * 2   // f16 codebook (padded)
                    + (size_t)KC * 4                // c_sq
                    + 8 * 32 * 4;                   // per-wave index staging
  vq_main<<<128, 256, smem, stream>>>(x, cb, cb16, csq, out);
}
